// CoAttention_7670811591148
// MI455X (gfx1250) — matrix operand order repaired and checked by reading, not compiled, not run on hardware
//
#include <hip/hip_runtime.h>

#define B8 8
#define B2 4
#define CH 256
#define CQ 64
#define HW 3136
#define M2 6272
#define W56 56
#define EPS 1e-5f

typedef __bf16 bf16_t;
typedef bf16_t bf16x16 __attribute__((ext_vector_type(16)));
typedef bf16_t bf16x8  __attribute__((ext_vector_type(8)));
typedef float  v8f     __attribute__((ext_vector_type(8)));

#define WMMA_BF16(a, b, c) \
  __builtin_amdgcn_wmma_f32_16x16x32_bf16(false, (a), false, (b), (short)0, (c), false, false)

// A-fragment (16x32 bf16, M = lane%16): regs0-7 hold K = k0+16h+0..15
__device__ __forceinline__ bf16x16 load_a_frag(const bf16_t* rowptr, int k0, int h) {
  bf16x8 lo = *(const bf16x8*)(rowptr + k0 + 16 * h);
  bf16x8 hi = *(const bf16x8*)(rowptr + k0 + 16 * h + 8);
  bf16x16 a;
#pragma unroll
  for (int i = 0; i < 8; ++i) { a[i] = lo[i]; a[i + 8] = hi[i]; }
  return a;
}

__device__ __forceinline__ v8f v8f_zero() {
  v8f z = {0.f, 0.f, 0.f, 0.f, 0.f, 0.f, 0.f, 0.f};
  return z;
}

// ---------------- gating: 3x3 conv (1 out channel) ----------------
__global__ void gate_conv(const float* __restrict__ t, const float* __restrict__ w_sa,
                          const float* __restrict__ b_sa, float* __restrict__ s_conv) {
  int idx = blockIdx.x * blockDim.x + threadIdx.x;
  if (idx >= B8 * HW) return;
  int b = idx / HW, p = idx % HW;
  int y = p / W56, x = p % W56;
  float acc = b_sa[0];
  for (int c = 0; c < CH; ++c) {
    const float* tc = t + ((size_t)b * CH + c) * HW;
    const float* wc = w_sa + c * 9;
#pragma unroll
    for (int dy = -1; dy <= 1; ++dy) {
      int yy = y + dy;
      if (yy < 0 || yy >= W56) continue;
#pragma unroll
      for (int dx = -1; dx <= 1; ++dx) {
        int xx = x + dx;
        if (xx < 0 || xx >= W56) continue;
        acc += tc[yy * W56 + xx] * wc[(dy + 1) * 3 + (dx + 1)];
      }
    }
  }
  s_conv[idx] = acc;
}

__global__ void gate_softmax(const float* __restrict__ s_conv, float* __restrict__ gate) {
  int idx = blockIdx.x * blockDim.x + threadIdx.x;
  if (idx >= B2 * HW) return;
  int pb = idx / HW, p = idx % HW;
  float s1 = s_conv[(size_t)pb * HW + p];
  float s2 = s_conv[(size_t)(pb + 4) * HW + p];
  float mx = fmaxf(s1, s2);
  float e1 = __expf(s1 - mx), e2 = __expf(s2 - mx);
  float inv = 1.0f / (e1 + e2);
  gate[(size_t)pb * HW + p] = e1 * inv;
  gate[(size_t)(pb + 4) * HW + p] = e2 * inv;
}

// ---------------- w_o -> bf16 ----------------
__global__ void cvt_wo(const float* __restrict__ w_o, bf16_t* __restrict__ wo_bf) {
  int i = blockIdx.x * blockDim.x + threadIdx.x;
  if (i < CH * 2 * CH) wo_bf[i] = (bf16_t)w_o[i];
}

// ---------------- q/k/v 1x1 projections -> bf16 + bf16 copy of t ----------------
// qb   [B2][HW][64]   (row n contiguous over kc)
// kTb  [B2][M2][64]   (row m contiguous over kc)
// vb   [B2][CH][M2]   (row c contiguous over m)
// t_bf [B8][CH][HW]
__global__ void proj_qkv(const float* __restrict__ t, const float* __restrict__ w_q,
                         const float* __restrict__ w_k, const float* __restrict__ w_v,
                         const float* __restrict__ g_v, const float* __restrict__ bt_v,
                         const float* __restrict__ m_v, const float* __restrict__ var_v,
                         bf16_t* __restrict__ qb, bf16_t* __restrict__ kTb,
                         bf16_t* __restrict__ vb, bf16_t* __restrict__ t_bf) {
  __shared__ float tl[CH][16];
  __shared__ float dl[CH][16];
  int b = blockIdx.y;
  int p0 = blockIdx.x * 16;
  int c = threadIdx.x;  // 256 threads
  const float* trow = t + ((size_t)b * CH + c) * HW + p0;
  float tv[16];
#pragma unroll
  for (int px = 0; px < 16; ++px) tv[px] = trow[px];
#pragma unroll
  for (int px = 0; px < 16; ++px) tl[c][px] = tv[px];
  {
    bf16_t* td = t_bf + ((size_t)b * CH + c) * HW + p0;
#pragma unroll
    for (int px = 0; px < 16; ++px) td[px] = (bf16_t)tv[px];
  }
  if (b < 4) {
    const float* trow2 = t + ((size_t)(b + 4) * CH + c) * HW + p0;
#pragma unroll
    for (int px = 0; px < 16; ++px) dl[c][px] = fabsf(tv[px] - trow2[px]);
  }
  __syncthreads();

  int pb = b & 3;
  size_t moff = (b < 4) ? 0 : (size_t)HW;

  // v = relu(bn(w_v @ t)), all 256 threads
  {
    float inv = g_v[c] * rsqrtf(var_v[c] + EPS);
    float bias = bt_v[c] - m_v[c] * inv;
    const float* wr = w_v + (size_t)c * CH;
    float acc[16];
#pragma unroll
    for (int px = 0; px < 16; ++px) acc[px] = 0.f;
    for (int i = 0; i < CH; ++i) {
      float w = wr[i];
#pragma unroll
      for (int px = 0; px < 16; ++px) acc[px] += w * tl[i][px];
    }
    bf16_t* dst = vb + ((size_t)pb * CH + c) * M2 + moff + p0;
#pragma unroll
    for (int px = 0; px < 16; ++px)
      dst[px] = (bf16_t)fmaxf(acc[px] * inv + bias, 0.f);
  }
  // k = w_k @ t, threads 0..63
  if (c < CQ) {
    const float* wr = w_k + (size_t)c * CH;
    float acc[16];
#pragma unroll
    for (int px = 0; px < 16; ++px) acc[px] = 0.f;
    for (int i = 0; i < CH; ++i) {
      float w = wr[i];
#pragma unroll
      for (int px = 0; px < 16; ++px) acc[px] += w * tl[i][px];
    }
#pragma unroll
    for (int px = 0; px < 16; ++px)
      kTb[((size_t)pb * M2 + moff + p0 + px) * CQ + c] = (bf16_t)acc[px];
  }
  // q = w_q @ |t1 - t2|, threads 64..127, only b < 4
  if (b < 4 && c >= 64 && c < 128) {
    int kc = c - 64;
    const float* wr = w_q + (size_t)kc * CH;
    float acc[16];
#pragma unroll
    for (int px = 0; px < 16; ++px) acc[px] = 0.f;
    for (int i = 0; i < CH; ++i) {
      float w = wr[i];
#pragma unroll
      for (int px = 0; px < 16; ++px) acc[px] += w * dl[i][px];
    }
#pragma unroll
    for (int px = 0; px < 16; ++px)
      qb[((size_t)pb * HW + p0 + px) * CQ + kc] = (bf16_t)acc[px];
  }
}

// ---------------- attention pass 1: softmax row stats ----------------
// Lane-private online (max,sum) with a SINGLE exp per element:
//   e = exp(-|s-mx|); s>mx ? (ls = ls*e + 1, mx = s) : (ls += e)
// Butterfly merge across the 16-lane half only AFTER the m-loop.
// Emits shift[n] = rowmax + ln(rowsum) so pass 2 does p = exp(s - shift).
__global__ void attn_stats(const bf16_t* __restrict__ qb, const bf16_t* __restrict__ kTb,
                           float* __restrict__ shift) {
  int wave = threadIdx.x >> 5;
  int lane = threadIdx.x & 31;
  int ln = lane & 15, h = lane >> 4;
  int pb = blockIdx.y;
  int n0 = (blockIdx.x * 4 + wave) * 16;

  const bf16_t* qrow = qb + ((size_t)pb * HW + n0 + ln) * CQ;
  const bf16_t* kbase = kTb + (size_t)pb * M2 * CQ;
  bf16x16 a0 = load_a_frag(qrow, 0, h);
  bf16x16 a1 = load_a_frag(qrow, 32, h);

  float mx[8], ls[8];
#pragma unroll
  for (int r = 0; r < 8; ++r) { mx[r] = -3.0e38f; ls[r] = 0.f; }

#pragma unroll 2
  for (int m0 = 0; m0 < M2; m0 += 16) {
    const bf16_t* krow = kbase + (size_t)(m0 + ln) * CQ;
    bf16x16 b0 = *(const bf16x16*)(krow + 16 * h);        // K = 0..31 half
    bf16x16 b1 = *(const bf16x16*)(krow + 32 + 16 * h);   // K = 32..63 half
    v8f s = v8f_zero();
    s = WMMA_BF16(a0, b0, s);
    s = WMMA_BF16(a1, b1, s);
#pragma unroll
    for (int r = 0; r < 8; ++r) {
      float d = s[r] - mx[r];
      float e = __expf(-__builtin_fabsf(d));
      bool gt = d > 0.f;
      ls[r] = gt ? __builtin_fmaf(ls[r], e, 1.f) : (ls[r] + e);
      mx[r] = gt ? s[r] : mx[r];
    }
  }
  // merge (max,sum) pairs across the 16 lanes of each half
#pragma unroll
  for (int off = 1; off < 16; off <<= 1) {
#pragma unroll
    for (int r = 0; r < 8; ++r) {
      float om = __shfl_xor(mx[r], off, 32);
      float os = __shfl_xor(ls[r], off, 32);
      float nm = fmaxf(mx[r], om);
      ls[r] = ls[r] * __expf(mx[r] - nm) + os * __expf(om - nm);
      mx[r] = nm;
    }
  }
  float* sb = shift + (size_t)pb * HW;
#pragma unroll
  for (int r = 0; r < 8; ++r) {
    if (ln == r) {
      int row = n0 + r + 8 * h;
      sb[row] = mx[r] + __logf(ls[r]);
    }
  }
}

// ---------------- attention pass 2: out(n,c) = sum_m softmax(s)(n,m) * v(c,m) ----------------
__global__ void attn_apply(const bf16_t* __restrict__ qb, const bf16_t* __restrict__ kTb,
                           const bf16_t* __restrict__ vb, const float* __restrict__ shift,
                           bf16_t* __restrict__ vatt) {
  __shared__ bf16_t Plds[4][16 * 32];
  int wave = threadIdx.x >> 5;
  int lane = threadIdx.x & 31;
  int ln = lane & 15, h = lane >> 4;
  int pb = blockIdx.y;
  int n0 = (blockIdx.x * 4 + wave) * 16;

  const bf16_t* qrow = qb + ((size_t)pb * HW + n0 + ln) * CQ;
  const bf16_t* kbase = kTb + (size_t)pb * M2 * CQ;
  const bf16_t* vbase = vb + (size_t)pb * CH * M2;
  bf16x16 a0 = load_a_frag(qrow, 0, h);
  bf16x16 a1 = load_a_frag(qrow, 32, h);

  float sh[8];
#pragma unroll
  for (int r = 0; r < 8; ++r)
    sh[r] = shift[(size_t)pb * HW + n0 + r + 8 * h];

  v8f acc[16];
#pragma unroll
  for (int ct = 0; ct < 16; ++ct) acc[ct] = v8f_zero();

  bf16_t* pt = Plds[wave];
  for (int m0 = 0; m0 < M2; m0 += 32) {
#pragma unroll
    for (int sub = 0; sub < 2; ++sub) {
      int mt = m0 + sub * 16;
      const bf16_t* krow = kbase + (size_t)(mt + ln) * CQ;
      bf16x16 b0 = *(const bf16x16*)(krow + 16 * h);
      bf16x16 b1 = *(const bf16x16*)(krow + 32 + 16 * h);
      v8f s = v8f_zero();
      s = WMMA_BF16(a0, b0, s);
      s = WMMA_BF16(a1, b1, s);
#pragma unroll
      for (int r = 0; r < 8; ++r) {
        float p = __expf(s[r] - sh[r]);  // normalization folded into shift
        pt[(r + 8 * h) * 32 + sub * 16 + ln] = (bf16_t)p;  // row-major 16x32 P tile
      }
    }
    // D-layout -> A-layout through LDS
    bf16x16 pa;
    {
      bf16x8 lo = *(const bf16x8*)(pt + ln * 32 + 16 * h);
      bf16x8 hi = *(const bf16x8*)(pt + ln * 32 + 16 * h + 8);
#pragma unroll
      for (int i = 0; i < 8; ++i) { pa[i] = lo[i]; pa[i + 8] = hi[i]; }
    }
#pragma unroll
    for (int ct = 0; ct < 16; ++ct) {
      const bf16_t* vr = vbase + (size_t)(ct * 16 + ln) * M2 + m0 + 16 * h;
      bf16x16 bv = *(const bf16x16*)vr;  // K = m contiguous, N = c = lane%16
      acc[ct] = WMMA_BF16(pa, bv, acc[ct]);
    }
  }
  // store vatt[pb][n][c] in bf16 (n-major so out_conv reads channel-contiguous)
  bf16_t* vo = vatt + (size_t)pb * HW * CH;
#pragma unroll
  for (int ct = 0; ct < 16; ++ct) {
#pragma unroll
    for (int r = 0; r < 8; ++r) {
      vo[(size_t)(n0 + r + 8 * h) * CH + ct * 16 + ln] = (bf16_t)acc[ct][r];
    }
  }
}

// ---------------- final: out = relu(bn(w_o @ concat(t, gate * v_att))) ----------------
__global__ void out_conv(const bf16_t* __restrict__ t_bf, const float* __restrict__ gate,
                         const bf16_t* __restrict__ vatt, const bf16_t* __restrict__ wo_bf,
                         const float* __restrict__ g_o, const float* __restrict__ bt_o,
                         const float* __restrict__ m_o, const float* __restrict__ var_o,
                         float* __restrict__ out) {
  int wave = threadIdx.x >> 5;
  int lane = threadIdx.x & 31;
  int ln = lane & 15, h = lane >> 4;
  int b = blockIdx.z;
  int ctile = blockIdx.y;
  int p0 = (blockIdx.x * 4 + wave) * 16;
  int pb = b & 3;
  int p = p0 + ln;

  const bf16_t* tb = t_bf + (size_t)b * CH * HW;
  float gv = gate[(size_t)b * HW + p];
  const bf16_t* vr = vatt + ((size_t)pb * HW + p) * CH;
  const bf16_t* wrow = wo_bf + (size_t)(ctile * 16 + ln) * (2 * CH);

  v8f acc = v8f_zero();
#pragma unroll
  for (int k0 = 0; k0 < 2 * CH; k0 += 32) {
    bf16x16 a = load_a_frag(wrow, k0, h);
    bf16x16 x;
    if (k0 < CH) {
      // B: t channels (K strided by HW, coalesced across lanes over pixels)
      const bf16_t* tc = tb + (size_t)(k0 + 16 * h) * HW + p;
#pragma unroll
      for (int i = 0; i < 16; ++i) x[i] = tc[(size_t)i * HW];
    } else {
      // B: gated attention output, channel-contiguous vector load
      bf16x16 vv = *(const bf16x16*)(vr + (k0 - CH) + 16 * h);
#pragma unroll
      for (int i = 0; i < 16; ++i) x[i] = (bf16_t)(gv * (float)vv[i]);
    }
    acc = WMMA_BF16(a, x, acc);
  }
  float* ob = out + (size_t)b * CH * HW;
#pragma unroll
  for (int r = 0; r < 8; ++r) {
    int co = ctile * 16 + r + 8 * h;
    float inv = g_o[co] * rsqrtf(var_o[co] + EPS);
    float bias = bt_o[co] - m_o[co] * inv;
    ob[(size_t)co * HW + p] = fmaxf(acc[r] * inv + bias, 0.f);
  }
}

extern "C" void kernel_launch(void* const* d_in, const int* in_sizes, int n_in,
                              void* d_out, int out_size, void* d_ws, size_t ws_size,
                              hipStream_t stream) {
  (void)in_sizes; (void)n_in; (void)out_size; (void)ws_size;
  const float* t     = (const float*)d_in[0];
  const float* w_sa  = (const float*)d_in[1];
  const float* b_sa  = (const float*)d_in[2];
  const float* w_q   = (const float*)d_in[3];
  const float* w_k   = (const float*)d_in[4];
  const float* w_v   = (const float*)d_in[5];
  const float* g_v   = (const float*)d_in[6];
  const float* bt_v  = (const float*)d_in[7];
  const float* m_v   = (const float*)d_in[8];
  const float* var_v = (const float*)d_in[9];
  const float* w_o   = (const float*)d_in[10];
  const float* g_o   = (const float*)d_in[11];
  const float* bt_o  = (const float*)d_in[12];
  const float* m_o   = (const float*)d_in[13];
  const float* var_o = (const float*)d_in[14];
  float* out = (float*)d_out;

  uintptr_t cur = ((uintptr_t)d_ws + 255) & ~(uintptr_t)255;
  auto take = [&cur](size_t bytes) {
    void* p = (void*)cur;
    cur = (cur + bytes + 255) & ~(uintptr_t)255;
    return p;
  };
  float*  s_conv = (float*)take((size_t)B8 * HW * sizeof(float));
  float*  gate   = (float*)take((size_t)B8 * HW * sizeof(float));
  bf16_t* qb     = (bf16_t*)take((size_t)B2 * HW * CQ * sizeof(bf16_t));
  bf16_t* kTb    = (bf16_t*)take((size_t)B2 * M2 * CQ * sizeof(bf16_t));
  bf16_t* vb     = (bf16_t*)take((size_t)B2 * CH * M2 * sizeof(bf16_t));
  float*  shift  = (float*)take((size_t)B2 * HW * sizeof(float));
  bf16_t* vatt   = (bf16_t*)take((size_t)B2 * HW * CH * sizeof(bf16_t));
  bf16_t* t_bf   = (bf16_t*)take((size_t)B8 * CH * HW * sizeof(bf16_t));
  bf16_t* wo_bf  = (bf16_t*)take((size_t)CH * 2 * CH * sizeof(bf16_t));

  gate_conv<<<(B8 * HW + 255) / 256, 256, 0, stream>>>(t, w_sa, b_sa, s_conv);
  gate_softmax<<<(B2 * HW + 255) / 256, 256, 0, stream>>>(s_conv, gate);
  cvt_wo<<<(CH * 2 * CH + 255) / 256, 256, 0, stream>>>(w_o, wo_bf);
  proj_qkv<<<dim3(HW / 16, B8), 256, 0, stream>>>(t, w_q, w_k, w_v, g_v, bt_v, m_v, var_v,
                                                  qb, kTb, vb, t_bf);
  attn_stats<<<dim3(HW / 64, B2), 128, 0, stream>>>(qb, kTb, shift);
  attn_apply<<<dim3(HW / 64, B2), 128, 0, stream>>>(qb, kTb, vb, shift, vatt);
  out_conv<<<dim3(HW / 64, CH / 16, B8), 128, 0, stream>>>(t_bf, gate, vatt, wo_bf, g_o, bt_o,
                                                           m_o, var_o, out);
}
